// model2_16801912062583
// MI455X (gfx1250) — matrix operand order repaired and checked by reading, not compiled, not run on hardware
//
#include <hip/hip_runtime.h>
#include <hip/hip_bf16.h>
#include <math.h>

typedef __attribute__((ext_vector_type(16))) __bf16 v16bf;
typedef __attribute__((ext_vector_type(8)))  float  v8f;

#define NTOT  27360
#define NNODE 1140
#define TSTEP 24
#define HDIM  256
#define KPAD  520            // 512 max K + 8 bf16 pad (1040B row stride -> good bank spread)

__device__ __forceinline__ float sigf(float x) { return 1.0f / (1.0f + __expf(-x)); }

// ---------------------------------------------------------------------------
// GEMM on WMMA bf16 -> f32.   C[M x Nc] = act(A[M x K] * B + bias(+bias2)(+addsrc))
// B element (k,n) = Bw[k*ldbk + n*ldbn].
// Block = 128 threads (4 waves). Block tile: 64 rows x 32 cols.
// Wave w owns rows [blkX*64 + w*16, +16) and BOTH 16-col sub-tiles (A reuse).
// B panel (kpad x 32) staged in LDS as bf16, zero-padded; K<=512 required.
// ---------------------------------------------------------------------------
union BFrag { v16bf v; uint4 q[2]; };

__global__ __launch_bounds__(128)
void wmma_gemm(const float* __restrict__ A, int lda,
               const float* __restrict__ Bw, int ldbk, int ldbn,
               const float* __restrict__ bias, const float* __restrict__ bias2,
               const float* __restrict__ addsrc, int addstride,
               float* __restrict__ C, int ldc,
               int M, int Nc, int K, int act, float slope)
{
    __shared__ __bf16 Bs[32 * KPAD];           // ~33 KB of 320 KB WGP LDS

    const int tid  = threadIdx.x;
    const int lane = tid & 31;
    const int wave = tid >> 5;
    const int half = lane >> 4;                // 0: lanes 0-15, 1: lanes 16-31
    const int lrow = lane & 15;
    const int m0   = blockIdx.x * 64 + wave * 16;
    const int n0   = blockIdx.y * 32;

    const int kpad  = (K + 31) & ~31;
    const int Kmain = K & ~31;

    // ---- cooperative B panel load: cols n0..n0+31, k rows 0..kpad-1 ----
    for (int idx = tid; idx < 32 * kpad; idx += 128) {
        int n = idx & 31, k = idx >> 5;
        int gc = n0 + n;
        float v = (gc < Nc && k < K) ? Bw[(size_t)k * ldbk + (size_t)gc * ldbn] : 0.0f;
        Bs[n * KPAD + k] = (__bf16)v;
    }
    __syncthreads();

    v8f acc0 = {0.f,0.f,0.f,0.f,0.f,0.f,0.f,0.f};
    v8f acc1 = {0.f,0.f,0.f,0.f,0.f,0.f,0.f,0.f};

    const int ar  = m0 + lrow;
    const int arc = (ar < M) ? ar : (M - 1);       // clamped: loads always in-bounds
    const float* __restrict__ Arow = A + (size_t)arc * lda;
    const __bf16* __restrict__ bp0 = &Bs[lrow * KPAD];          // col sub-tile 0
    const __bf16* __restrict__ bp1 = &Bs[(16 + lrow) * KPAD];   // col sub-tile 1

    // ---- main K loop: completely branch/check free ----
    for (int k0 = 0; k0 < Kmain; k0 += 32) {
        v16bf af;
        const float* pa = Arow + k0 + half * 8;
#pragma unroll
        for (int j = 0; j < 8; ++j) af[j]     = (__bf16)pa[j];
#pragma unroll
        for (int j = 0; j < 8; ++j) af[8 + j] = (__bf16)pa[16 + j];

        const __bf16* b0p = bp0 + k0 + half * 8;      // 16B-aligned (KPAD even, offs mult of 8)
        const __bf16* b1p = bp1 + k0 + half * 8;
        BFrag b0, b1;
        b0.q[0] = *(const uint4*)(b0p);     b0.q[1] = *(const uint4*)(b0p + 16);
        b1.q[0] = *(const uint4*)(b1p);     b1.q[1] = *(const uint4*)(b1p + 16);

        acc0 = __builtin_amdgcn_wmma_f32_16x16x32_bf16(false, af, false, b0.v, (short)0, acc0, false, false);
        acc1 = __builtin_amdgcn_wmma_f32_16x16x32_bf16(false, af, false, b1.v, (short)0, acc1, false, false);
    }

    // ---- single K tail iteration: cndmask selects, no exec branching ----
    if (Kmain < kpad) {
        const int k0 = Kmain;
        v16bf af;
#pragma unroll
        for (int j = 0; j < 16; ++j) {
            int kk = k0 + ((j < 8) ? (half * 8 + j) : (16 + half * 8 + (j - 8)));
            int kc = (kk < K) ? kk : (K - 1);
            float v = Arow[kc];
            af[j] = (__bf16)((kk < K) ? v : 0.0f);
        }
        const __bf16* b0p = bp0 + k0 + half * 8;      // B zero-padded to kpad
        const __bf16* b1p = bp1 + k0 + half * 8;
        BFrag b0, b1;
        b0.q[0] = *(const uint4*)(b0p);     b0.q[1] = *(const uint4*)(b0p + 16);
        b1.q[0] = *(const uint4*)(b1p);     b1.q[1] = *(const uint4*)(b1p + 16);
        acc0 = __builtin_amdgcn_wmma_f32_16x16x32_bf16(false, af, false, b0.v, (short)0, acc0, false, false);
        acc1 = __builtin_amdgcn_wmma_f32_16x16x32_bf16(false, af, false, b1.v, (short)0, acc1, false, false);
    }

    // ---- epilogue ----
    const int cc0 = n0 + lrow;
    const int cc1 = n0 + 16 + lrow;

    auto post = [&](float v, int cc, int rr) -> float {
        if (bias)   v += bias[cc];
        if (bias2)  v += bias2[cc];
        if (addsrc) v += addsrc[(size_t)rr * addstride + cc];
        if (act == 1)      v = tanhf(v);
        else if (act == 2) v = (v > 0.0f) ? v : slope * v;
        return v;
    };

    if (m0 + 16 <= M) {                        // all 16 rows valid (common path)
        if (cc0 < Nc) {
#pragma unroll
            for (int r = 0; r < 8; ++r) {
                int rr = m0 + half * 8 + r;
                C[(size_t)rr * ldc + cc0] = post(acc0[r], cc0, rr);
            }
        }
        if (cc1 < Nc) {
#pragma unroll
            for (int r = 0; r < 8; ++r) {
                int rr = m0 + half * 8 + r;
                C[(size_t)rr * ldc + cc1] = post(acc1[r], cc1, rr);
            }
        }
    } else {
#pragma unroll
        for (int r = 0; r < 8; ++r) {
            int rr = m0 + half * 8 + r;
            if (rr < M && cc0 < Nc) C[(size_t)rr * ldc + cc0] = post(acc0[r], cc0, rr);
            if (rr < M && cc1 < Nc) C[(size_t)rr * ldc + cc1] = post(acc1[r], cc1, rr);
        }
    }
}

// ---------------------------------------------------------------------------
__global__ void fill_kernel(float* p, float v, int n) {
    int i = blockIdx.x * blockDim.x + threadIdx.x;
    if (i < n) p[i] = v;
}

__global__ void node_dot2(const float* __restrict__ h, const float* __restrict__ al,
                          const float* __restrict__ ar, float* el, float* er, int n, int d) {
    int i = blockIdx.x * blockDim.x + threadIdx.x;
    if (i >= n) return;
    float a = 0.f, b = 0.f;
    for (int c = 0; c < d; ++c) {
        float v = h[(size_t)i * d + c];
        a += v * al[c];
        b += v * ar[c];
    }
    el[i] = a; er[i] = b;
}

// ordered-int float atomic max (dest pre-filled with -3e38)
__device__ __forceinline__ void atomicMaxF(float* addr, float val) {
    if (val >= 0.0f) atomicMax((int*)addr, __float_as_int(val));
    else             atomicMin((unsigned int*)addr, __float_as_uint(val));
}

__global__ void gat_edge_logit(const float* __restrict__ el, const float* __restrict__ er,
                               const int* __restrict__ src, const int* __restrict__ dst,
                               float* ebuf, float* mbuf, int E) {
    int e = blockIdx.x * blockDim.x + threadIdx.x;
    if (e >= E) return;
    float v = el[src[e]] + er[dst[e]];
    v = (v > 0.0f) ? v : 0.2f * v;          // LeakyReLU(0.2)
    ebuf[e] = v;
    atomicMaxF(&mbuf[dst[e]], v);
}

__global__ void gat_edge_exp(float* ebuf, const float* __restrict__ mbuf,
                             float* den, const int* __restrict__ dst, int E) {
    int e = blockIdx.x * blockDim.x + threadIdx.x;
    if (e >= E) return;
    int d = dst[e];
    float ex = __expf(ebuf[e] - mbuf[d]);
    ebuf[e] = ex;
    atomicAdd(&den[d], ex);
}

__global__ void gat_edge_aggr(const float* __restrict__ ebuf, const float* __restrict__ h,
                              const int* __restrict__ src, const int* __restrict__ dst,
                              float* acc, int E, int d) {
    int e = blockIdx.x * blockDim.x + threadIdx.x;
    if (e >= E) return;
    float w = ebuf[e];
    int s = src[e], t = dst[e];
    for (int c = 0; c < d; ++c)
        atomicAdd(&acc[(size_t)t * d + c], h[(size_t)s * d + c] * w);
}

__global__ void gat_finalize(const float* __restrict__ acc, const float* __restrict__ den,
                             const float* __restrict__ b, float* out, int n, int d) {
    int idx = blockIdx.x * blockDim.x + threadIdx.x;
    if (idx >= n * d) return;
    int i = idx / d, c = idx % d;
    out[idx] = tanhf(acc[idx] / den[i] + b[c]);
}

__global__ void deg_count(const int* __restrict__ dst, float* deg, int E) {
    int e = blockIdx.x * blockDim.x + threadIdx.x;
    if (e < E) atomicAdd(&deg[dst[e]], 1.0f);
}

__global__ void sage_edge_aggr(const float* __restrict__ xn, const int* __restrict__ src,
                               const int* __restrict__ dst, float* acc, int E, int d) {
    int e = blockIdx.x * blockDim.x + threadIdx.x;
    if (e >= E) return;
    int s = src[e], t = dst[e];
    for (int c = 0; c < d; ++c)
        atomicAdd(&acc[(size_t)t * d + c], xn[(size_t)s * d + c]);
}

__global__ void sage_finalize(const float* __restrict__ selfp, const float* __restrict__ acc,
                              const float* __restrict__ deg, const float* __restrict__ b,
                              float* out, int n, int d) {
    int idx = blockIdx.x * blockDim.x + threadIdx.x;
    if (idx >= n * d) return;
    int i = idx / d, c = idx % d;
    out[idx] = tanhf(selfp[idx] + acc[idx] / deg[i] + b[c]);
}

__global__ void concat47(const float* __restrict__ x, const float* __restrict__ g8, float* x47) {
    int idx = blockIdx.x * blockDim.x + threadIdx.x;
    if (idx >= NTOT * 47) return;
    int i = idx / 47, c = idx % 47;
    x47[idx] = (c < 39) ? x[(size_t)i * 39 + c] : g8[(size_t)i * 8 + (c - 39)];
}

__global__ void concat55(const float* __restrict__ x47, const float* __restrict__ s8, float* x55) {
    int idx = blockIdx.x * blockDim.x + threadIdx.x;
    if (idx >= NTOT * 55) return;
    int i = idx / 55, c = idx % 55;
    x55[idx] = (c < 47) ? x47[(size_t)i * 47 + c] : s8[(size_t)i * 8 + (c - 47)];
}

__global__ void lstm_gate(const float* __restrict__ g, float* c, float* h, float* hs, int t) {
    int idx = blockIdx.x * blockDim.x + threadIdx.x;
    if (idx >= NNODE * HDIM) return;
    int n = idx / HDIM, j = idx % HDIM;
    const float* gr = g + (size_t)n * 1024;
    float iv = sigf(gr[j]);
    float fv = sigf(gr[256 + j]);
    float gv = tanhf(gr[512 + j]);
    float ov = sigf(gr[768 + j]);
    float cv = fv * c[idx] + iv * gv;
    c[idx] = cv;
    float hv = ov * tanhf(cv);
    h[idx] = hv;
    hs[((size_t)n * TSTEP + t) * HDIM + j] = hv;
}

// z[r=n*24+t][c] : c<47 -> x47[node=t*N+n], else hs[r][c-47]
__global__ void concat_z(const float* __restrict__ x47, const float* __restrict__ hs, float* z) {
    int idx = blockIdx.x * blockDim.x + threadIdx.x;
    if (idx >= NTOT * 303) return;
    int r = idx / 303, c = idx % 303;
    if (c < 47) {
        int t = r % 24, n = r / 24;
        z[idx] = x47[((size_t)t * NNODE + n) * 47 + c];
    } else {
        z[idx] = hs[(size_t)r * 256 + (c - 47)];
    }
}

// BatchNorm1d over channel = t (B=24), normalized over (n, c); in-place
__global__ void bn_kernel(float* z, const float* __restrict__ gam, const float* __restrict__ bet) {
    __shared__ float ssum[256], ssq[256];
    const int ch = blockIdx.x;          // 0..23
    const int tid = threadIdx.x;
    const int CNT = NNODE * 303;
    float s = 0.f, q = 0.f;
    for (int e = tid; e < CNT; e += blockDim.x) {
        int n = e / 303, c = e % 303;
        float v = z[((size_t)n * 24 + ch) * 303 + c];
        s += v; q += v * v;
    }
    ssum[tid] = s; ssq[tid] = q; __syncthreads();
    for (int st = 128; st > 0; st >>= 1) {
        if (tid < st) { ssum[tid] += ssum[tid + st]; ssq[tid] += ssq[tid + st]; }
        __syncthreads();
    }
    float mean = ssum[0] / (float)CNT;
    float var  = ssq[0] / (float)CNT - mean * mean;
    float sc = gam[ch] * rsqrtf(var + 1e-3f);
    float sh = bet[ch] - mean * sc;
    for (int e = tid; e < CNT; e += blockDim.x) {
        int n = e / 303, c = e % 303;
        size_t id = ((size_t)n * 24 + ch) * 303 + c;
        z[id] = z[id] * sc + sh;
    }
}

// last MLP layer 80->1, fused [N,B] -> [B,N] transpose on store
__global__ void final_out(const float* __restrict__ a, const float* __restrict__ w,
                          const float* __restrict__ b, float* out) {
    int r = blockIdx.x * blockDim.x + threadIdx.x;
    if (r >= NTOT) return;
    float s = b[0];
    for (int c = 0; c < 80; ++c) s += a[(size_t)r * 80 + c] * w[c];
    int t = r % 24, n = r / 24;
    out[(size_t)t * NNODE + n] = s;
}

// ---------------------------------------------------------------------------
extern "C" void kernel_launch(void* const* d_in, const int* in_sizes, int n_in,
                              void* d_out, int out_size, void* d_ws, size_t ws_size,
                              hipStream_t stream)
{
    const float* x   = (const float*)d_in[0];
    const int*   src = (const int*)d_in[1];
    const int*   dst = (const int*)d_in[2];
    const int E = in_sizes[1];

    const float* gW[3]  = {(const float*)d_in[3],  (const float*)d_in[7],  (const float*)d_in[11]};
    const float* gal[3] = {(const float*)d_in[4],  (const float*)d_in[8],  (const float*)d_in[12]};
    const float* gar[3] = {(const float*)d_in[5],  (const float*)d_in[9],  (const float*)d_in[13]};
    const float* gb[3]  = {(const float*)d_in[6],  (const float*)d_in[10], (const float*)d_in[14]};
    const float* sWs[3] = {(const float*)d_in[15], (const float*)d_in[18], (const float*)d_in[21]};
    const float* sWn[3] = {(const float*)d_in[16], (const float*)d_in[19], (const float*)d_in[22]};
    const float* sb[3]  = {(const float*)d_in[17], (const float*)d_in[20], (const float*)d_in[23]};
    const float* Wih = (const float*)d_in[24];   // [1024,55]
    const float* Whh = (const float*)d_in[25];   // [1024,256]
    const float* bih = (const float*)d_in[26];
    const float* bhh = (const float*)d_in[27];
    const float* bng = (const float*)d_in[28];
    const float* bnb = (const float*)d_in[29];
    const float* mW[5] = {(const float*)d_in[30], (const float*)d_in[32], (const float*)d_in[34],
                          (const float*)d_in[36], (const float*)d_in[38]};
    const float* mb[5] = {(const float*)d_in[31], (const float*)d_in[33], (const float*)d_in[35],
                          (const float*)d_in[37], (const float*)d_in[39]};

    // ---- workspace layout (floats) -------------------------------------
    float* ws = (float*)d_ws;
    float* Z    = ws;                              // NT*303 (persists)
    float* P    = Z + (size_t)NTOT * 303;          // reused pool
    // phase A (GNN + LSTM):
    float* HS   = P;                               // NT*256
    float* X47  = HS  + (size_t)NTOT * 256;        // NT*47
    float* H32  = X47 + (size_t)NTOT * 47;         // NT*32
    float* ACC  = H32 + (size_t)NTOT * 32;
    float* GOUT = ACC + (size_t)NTOT * 32;
    float* SELF = GOUT + (size_t)NTOT * 32;
    float* SOUT = SELF + (size_t)NTOT * 32;
    float* EL   = SOUT + (size_t)NTOT * 32;
    float* ER   = EL  + NTOT;
    float* MB   = ER  + NTOT;
    float* DEN  = MB  + NTOT;
    float* DEG  = DEN + NTOT;
    float* EBUF = DEG + NTOT;                      // E
    float* XSEQ = EBUF + (size_t)E;                // NT*55
    float* G    = XSEQ + (size_t)NTOT * 55;        // 1140*1024
    float* HC   = G   + (size_t)NNODE * 1024;      // 1140*256
    float* CC   = HC  + (size_t)NNODE * 256;
    // phase B (MLP) reuses pool:
    float* M1 = P;                                 // NT*512
    float* M2 = P + (size_t)NTOT * 512;            // NT*320

    auto gemm = [&](const float* A, int lda, const float* Bw, int ldbk, int ldbn,
                    const float* bias, const float* bias2,
                    const float* addsrc, int addstride,
                    float* C, int ldc, int M, int Nc, int K, int act, float slope) {
        dim3 g((M + 63) / 64, (Nc + 31) / 32);
        hipLaunchKernelGGL(wmma_gemm, g, dim3(128), 0, stream,
                           A, lda, Bw, ldbk, ldbn, bias, bias2, addsrc, addstride,
                           C, ldc, M, Nc, K, act, slope);
    };
    const int TB = 256;
    auto blocks = [&](int n) { return (n + TB - 1) / TB; };

    // ================= GAT x3 =================
    const float* gin = x; int gdi = 39;
    const int gdo[3] = {32, 16, 8};
    for (int l = 0; l < 3; ++l) {
        int d = gdo[l];
        gemm(gin, gdi, gW[l], d, 1, nullptr, nullptr, nullptr, 0,
             H32, d, NTOT, d, gdi, 0, 0.f);
        hipLaunchKernelGGL(node_dot2, dim3(blocks(NTOT)), dim3(TB), 0, stream,
                           H32, gal[l], gar[l], EL, ER, NTOT, d);
        hipLaunchKernelGGL(fill_kernel, dim3(blocks(NTOT)), dim3(TB), 0, stream,
                           MB, -3.0e38f, NTOT);
        hipMemsetAsync(DEN, 0, (size_t)NTOT * sizeof(float), stream);
        hipMemsetAsync(ACC, 0, (size_t)NTOT * d * sizeof(float), stream);
        hipLaunchKernelGGL(gat_edge_logit, dim3(blocks(E)), dim3(TB), 0, stream,
                           EL, ER, src, dst, EBUF, MB, E);
        hipLaunchKernelGGL(gat_edge_exp, dim3(blocks(E)), dim3(TB), 0, stream,
                           EBUF, MB, DEN, dst, E);
        hipLaunchKernelGGL(gat_edge_aggr, dim3(blocks(E)), dim3(TB), 0, stream,
                           EBUF, H32, src, dst, ACC, E, d);
        hipLaunchKernelGGL(gat_finalize, dim3(blocks(NTOT * d)), dim3(TB), 0, stream,
                           ACC, DEN, gb[l], GOUT, NTOT, d);
        gin = GOUT; gdi = d;
    }

    // x47 = concat(x, gat_out8)
    hipLaunchKernelGGL(concat47, dim3(blocks(NTOT * 47)), dim3(TB), 0, stream, x, GOUT, X47);

    // degree (includes self loops)
    hipMemsetAsync(DEG, 0, (size_t)NTOT * sizeof(float), stream);
    hipLaunchKernelGGL(deg_count, dim3(blocks(E)), dim3(TB), 0, stream, dst, DEG, E);

    // ================= SAGE x3 =================
    const float* sin_ = X47; int sdi = 47;
    const int sdo[3] = {32, 16, 8};
    for (int l = 0; l < 3; ++l) {
        int d = sdo[l];
        gemm(sin_, sdi, sWs[l], d, 1, nullptr, nullptr, nullptr, 0,
             SELF, d, NTOT, d, sdi, 0, 0.f);
        gemm(sin_, sdi, sWn[l], d, 1, nullptr, nullptr, nullptr, 0,
             H32, d, NTOT, d, sdi, 0, 0.f);
        hipMemsetAsync(ACC, 0, (size_t)NTOT * d * sizeof(float), stream);
        hipLaunchKernelGGL(sage_edge_aggr, dim3(blocks(E)), dim3(TB), 0, stream,
                           H32, src, dst, ACC, E, d);
        hipLaunchKernelGGL(sage_finalize, dim3(blocks(NTOT * d)), dim3(TB), 0, stream,
                           SELF, ACC, DEG, sb[l], SOUT, NTOT, d);
        sin_ = SOUT; sdi = d;
    }

    // x55 sequence buffer (node order == [t][n] contiguous)
    hipLaunchKernelGGL(concat55, dim3(blocks(NTOT * 55)), dim3(TB), 0, stream, X47, SOUT, XSEQ);

    // ================= LSTM (T=24 steps) =================
    hipMemsetAsync(HC, 0, (size_t)NNODE * 256 * sizeof(float), stream);
    hipMemsetAsync(CC, 0, (size_t)NNODE * 256 * sizeof(float), stream);
    for (int t = 0; t < TSTEP; ++t) {
        // g = x55_t @ Wih^T + bih + bhh       (Wih [1024,55] -> B(k,n)=Wih[n*55+k])
        gemm(XSEQ + (size_t)t * NNODE * 55, 55, Wih, 1, 55, bih, bhh, nullptr, 0,
             G, 1024, NNODE, 1024, 55, 0, 0.f);
        // g += h @ Whh^T                      (Whh [1024,256] -> B(k,n)=Whh[n*256+k])
        gemm(HC, 256, Whh, 1, 256, nullptr, nullptr, G, 1024,
             G, 1024, NNODE, 1024, 256, 0, 0.f);
        hipLaunchKernelGGL(lstm_gate, dim3(blocks(NNODE * HDIM)), dim3(TB), 0, stream,
                           G, CC, HC, HS, t);
    }

    // ================= z concat + BN =================
    hipLaunchKernelGGL(concat_z, dim3(blocks(NTOT * 303)), dim3(TB), 0, stream, X47, HS, Z);
    hipLaunchKernelGGL(bn_kernel, dim3(24), dim3(256), 0, stream, Z, bng, bnb);

    // ================= MLP =================
    gemm(Z,  303, mW[0], 512, 1, mb[0], nullptr, nullptr, 0, M1, 512, NTOT, 512, 303, 2, 0.01f);
    gemm(M1, 512, mW[1], 320, 1, mb[1], nullptr, nullptr, 0, M2, 320, NTOT, 320, 512, 2, 0.01f);
    gemm(M2, 320, mW[2], 160, 1, mb[2], nullptr, nullptr, 0, M1, 160, NTOT, 160, 320, 2, 0.01f);
    gemm(M1, 160, mW[3],  80, 1, mb[3], nullptr, nullptr, 0, M2,  80, NTOT,  80, 160, 2, 0.01f);
    hipLaunchKernelGGL(final_out, dim3(blocks(NTOT)), dim3(TB), 0, stream,
                       M2, mW[4], mb[4], (float*)d_out);
}
